// RNNClassifier_30451318129216
// MI455X (gfx1250) — matrix operand order repaired and checked by reading, not compiled, not run on hardware
//
#include <hip/hip_runtime.h>
#include <hip/hip_bf16.h>

// ---------------------------------------------------------------------------
// Types / helpers
// ---------------------------------------------------------------------------
typedef __attribute__((ext_vector_type(16))) __bf16 v16bf;
typedef __attribute__((ext_vector_type(8)))  float  v8f;

union Frag16 { unsigned short u[16]; unsigned x[8]; v16bf v; };

__device__ inline unsigned short f32_to_bf16(float f) {
    __bf16 h = (__bf16)f;                       // native cvt, RNE
    return __builtin_bit_cast(unsigned short, h);
}

__device__ inline v16bf frag_from(uint4 lo, uint4 hi) {
    Frag16 r;
    r.x[0] = lo.x; r.x[1] = lo.y; r.x[2] = lo.z; r.x[3] = lo.w;
    r.x[4] = hi.x; r.x[5] = hi.y; r.x[6] = hi.z; r.x[7] = hi.w;
    return r.v;
}

// A-fragment (16x32 bf16) from row-major bf16 LDS [m][LDA]:
// lane needs ushorts [m][kb..kb+15] -> two b128 loads.
// (vgpr i -> K = lanehi*16 + i*2)
template<int LDA>
__device__ inline v16bf load_a_frag(const unsigned short* S, int m0, int k0, int lane) {
    int m  = m0 + (lane & 15);
    int kb = k0 + ((lane & 16) ? 16 : 0);
    uint4 lo = *(const uint4*)&S[m * LDA + kb];
    uint4 hi = *(const uint4*)&S[m * LDA + kb + 8];
    return frag_from(lo, hi);
}

// B-fragment (32x16 bf16) from swizzled storage:
// per (k-tile, n-subtile) block of 256 uints laid out [nl(16)][lanehi(2)][pi(8)],
// uint(pi) = {bf16(K=lh*16+2*pi, n), bf16(K+1, n)}  -> two b128 loads per lane.
__device__ inline v16bf load_b_frag_sw(const unsigned* S, int subtile, int lane) {
    const unsigned* base = S + subtile * 256 + (lane & 15) * 16 + ((lane & 16) ? 8 : 0);
    uint4 lo = *(const uint4*)(base);
    uint4 hi = *(const uint4*)(base + 4);
    return frag_from(lo, hi);
}

#define NBLK  16
#define SLICE 32

__device__ inline void grid_sync(unsigned* s) {
    __syncthreads();
    if (threadIdx.x == 0) {
        __threadfence();
        unsigned gen = __hip_atomic_load(&s[1], __ATOMIC_ACQUIRE, __HIP_MEMORY_SCOPE_AGENT);
        unsigned arrived = __hip_atomic_fetch_add(&s[0], 1u, __ATOMIC_ACQ_REL, __HIP_MEMORY_SCOPE_AGENT);
        if (arrived == NBLK - 1) {
            __hip_atomic_store(&s[0], 0u, __ATOMIC_RELAXED, __HIP_MEMORY_SCOPE_AGENT);
            __hip_atomic_fetch_add(&s[1], 1u, __ATOMIC_ACQ_REL, __HIP_MEMORY_SCOPE_AGENT);
        } else {
            while (__hip_atomic_load(&s[1], __ATOMIC_ACQUIRE, __HIP_MEMORY_SCOPE_AGENT) == gen)
                __builtin_amdgcn_s_sleep(2);
        }
    }
    __syncthreads();
}

// ---------------------------------------------------------------------------
// Setup kernels
// ---------------------------------------------------------------------------
__global__ void init_sync_kernel(unsigned* s) { s[0] = 0u; s[1] = 0u; }

// f32 [K][Nsrc] -> swizzled K-pair bf16x2 uints in a matrix of Ntot columns:
// out[((kt*(Ntot/16) + ns)*16 + nl)*16 + lh*8 + pi], k = kt*32 + lh*16 + pi*2,
// global col = colOff + n, ns = gcol/16, nl = gcol%16.
__global__ void cvt_swz_kernel(const float* __restrict__ src, unsigned* __restrict__ dst,
                               int K2, int Nsrc, int Ntot, int colOff) {
    int i = blockIdx.x * 256 + threadIdx.x;
    if (i < K2 * Nsrc) {
        int k2 = i / Nsrc, n = i - k2 * Nsrc;
        int k  = 2 * k2;
        int kt = k2 >> 4;
        int pt = k2 & 15;            // pair row within k-tile
        int lh = pt >> 3, pi = pt & 7;
        int gcol = colOff + n;
        int ns = gcol >> 4, nl = gcol & 15;
        unsigned lo = f32_to_bf16(src[(size_t)k       * Nsrc + n]);
        unsigned hi = f32_to_bf16(src[(size_t)(k + 1) * Nsrc + n]);
        dst[(((size_t)kt * (Ntot >> 4) + ns) * 16 + nl) * 16 + lh * 8 + pi] = lo | (hi << 16);
    }
}

__global__ void copy_f32_kernel(const float* __restrict__ src, float* __restrict__ dst, int n) {
    int i = blockIdx.x * 256 + threadIdx.x;
    if (i < n) dst[i] = src[i];
}

// ---------------------------------------------------------------------------
// Tiled WMMA GEMM:  C[M,N] = (A + A2?)[M,K] @ Bswz[K,N] + bias[N]
// 128x128 block tile, BK=32, 8 waves; wave = 32 rows x 64 cols (8 wmma tiles).
// ---------------------------------------------------------------------------
template<bool HAS_A2>
__global__ __launch_bounds__(256) void gemm_bias(
    const float* __restrict__ A, const float* __restrict__ A2,
    const unsigned* __restrict__ Bw, const float* __restrict__ bias,
    float* __restrict__ C, int N, int K)
{
    __shared__ unsigned short As[128 * 32];   // 8 KB row-major [m][k]
    __shared__ unsigned      Bs[8 * 256];     // 8 KB: 8 n-subtiles x 256 uints

    const int tid  = threadIdx.x;
    const int lane = tid & 31;
    const int wave = tid >> 5;
    const int wm   = wave & 3;        // 4 m-groups of 32 rows
    const int wn   = wave >> 2;       // 2 n-groups of 64 cols (4 subtiles)
    const int row0 = blockIdx.y * 128;
    const int n0   = blockIdx.x * 128;
    const int nst  = N >> 4;          // n-subtiles per k-tile row of Bw

    v8f acc[2][4] = {};

    const int ar = tid >> 1;          // 0..127
    const int ac = (tid & 1) * 16;    // 0 or 16

    for (int k0 = 0; k0 < K; k0 += 32) {
        // A tile: 128 x 32 f32 -> bf16 (row-major)
        const float* ap = A + (size_t)(row0 + ar) * K + k0 + ac;
        if (k0 + 32 < K) __builtin_prefetch(ap + 32, 0, 0);   // next A k-tile
        float av[16];
        *(float4*)&av[0]  = *(const float4*)(ap);
        *(float4*)&av[4]  = *(const float4*)(ap + 4);
        *(float4*)&av[8]  = *(const float4*)(ap + 8);
        *(float4*)&av[12] = *(const float4*)(ap + 12);
        if (HAS_A2) {
            const float* ap2 = A2 + (size_t)(row0 + ar) * K + k0 + ac;
            if (k0 + 32 < K) __builtin_prefetch(ap2 + 32, 0, 0);
            float4 q0 = *(const float4*)(ap2);
            float4 q1 = *(const float4*)(ap2 + 4);
            float4 q2 = *(const float4*)(ap2 + 8);
            float4 q3 = *(const float4*)(ap2 + 12);
            av[0]+=q0.x; av[1]+=q0.y; av[2]+=q0.z; av[3]+=q0.w;
            av[4]+=q1.x; av[5]+=q1.y; av[6]+=q1.z; av[7]+=q1.w;
            av[8]+=q2.x; av[9]+=q2.y; av[10]+=q2.z; av[11]+=q2.w;
            av[12]+=q3.x; av[13]+=q3.y; av[14]+=q3.z; av[15]+=q3.w;
        }
#pragma unroll
        for (int i = 0; i < 16; ++i) As[ar * 32 + ac + i] = f32_to_bf16(av[i]);

        // B tile: contiguous 2048 uints (8 subtiles of this k-tile)
        {
            const unsigned* bp = Bw + ((size_t)(k0 >> 5) * nst + (n0 >> 4)) * 256 + tid * 8;
            *(uint4*)&Bs[tid * 8]     = *(const uint4*)(bp);
            *(uint4*)&Bs[tid * 8 + 4] = *(const uint4*)(bp + 4);
        }
        __syncthreads();

        v16bf a0 = load_a_frag<32>(As, wm * 32,      0, lane);
        v16bf a1 = load_a_frag<32>(As, wm * 32 + 16, 0, lane);
#pragma unroll
        for (int j = 0; j < 4; ++j) {
            v16bf b = load_b_frag_sw(Bs, wn * 4 + j, lane);
            acc[0][j] = __builtin_amdgcn_wmma_f32_16x16x32_bf16(false, a0, false, b, (short)0, acc[0][j], false, false);
            acc[1][j] = __builtin_amdgcn_wmma_f32_16x16x32_bf16(false, a1, false, b, (short)0, acc[1][j], false, false);
        }
        __syncthreads();
    }

    const int mr = row0 + wm * 32 + ((lane >> 4) << 3);
    const int nc = n0 + wn * 64 + (lane & 15);
#pragma unroll
    for (int j = 0; j < 4; ++j) {
        const float bia = bias[nc + 16 * j];
#pragma unroll
        for (int v = 0; v < 8; ++v) {
            C[(size_t)(mr + v)      * N + nc + 16 * j] = acc[0][j][v] + bia;
            C[(size_t)(mr + 16 + v) * N + nc + 16 * j] = acc[1][j][v] + bia;
        }
    }
}

// ---------------------------------------------------------------------------
// Persistent bf16-WMMA LSTM recurrence (16 blocks, slice of 32 hidden units).
// U slice LDS-resident in swizzled layout [kt(16)][ls(8)][256 uints].
// Z is the fused projection buffer [B*T][4096]; colBase selects direction.
// h_{t-1} staged global->LDS via GLOBAL_LOAD_ASYNC_TO_LDS_B128 (ASYNCcnt).
// ---------------------------------------------------------------------------
__global__ __launch_bounds__(256) void lstm_rec(
    const float* __restrict__ Z,           // [B*T][4096], row = b*1024+t
    const unsigned* __restrict__ Up,       // swizzled uints, Ntot=2048
    float* __restrict__ Hout,              // [B][T][1024]
    unsigned short* __restrict__ hprev,    // bf16 [64][512]
    unsigned* __restrict__ sync,
    int reverse, int dirOff, int colBase)
{
    __shared__ unsigned       Us[16 * 8 * 256]; // 128 KB U slice
    __shared__ unsigned short Hs[64 * 512];     //  64 KB h_{t-1} (row-major)
    __shared__ float          ZsT[128 * 64];    //  32 KB gate pre-acts, [n][m]
    __shared__ float          Cs[64 * SLICE];   //   8 KB cell state

    const int tid  = threadIdx.x;
    const int lane = tid & 31;
    const int wave = tid >> 5;
    const int wm   = wave & 3;
    const int wn   = wave >> 2;
    const int j0   = blockIdx.x * SLICE;

    // Preload U slice. Local subtile ls = gate*2 + half covers global subtile
    // ns = gate*32 + j0/16 + half (Ntot=2048 -> 128 subtiles per k-tile).
    for (int i = tid; i < 16 * 8 * 256; i += 256) {
        int kt = i >> 11;
        int rem = i & 2047;
        int ls = rem >> 8, w = rem & 255;
        int gate = ls >> 1;
        int nsg  = gate * 32 + (j0 >> 4) + (ls & 1);
        Us[i] = Up[((size_t)kt * 128 + nsg) * 256 + w];
    }
    for (int i = tid; i < 64 * SLICE; i += 256) {
        Cs[i] = 0.f;
        int b = i >> 5, j = i & 31;
        hprev[b * 512 + j0 + j] = 0;
    }
    grid_sync(sync);

    // Per-thread async-copy addresses for the 64 KB h_{t-1} staging:
    // 4096 x 16B chunks, 256 threads -> 16 chunks each.
    const unsigned ldsBase = (unsigned)(size_t)(&Hs[0]);   // LDS byte offset (addr[31:0])
    const char*    gBase   = (const char*)hprev;

    for (int s = 0; s < 1024; ++s) {
        const int t = reverse ? (1023 - s) : s;

        {   // stage h_{t-1} via async DMA to LDS (tracked by ASYNCcnt)
#pragma unroll
            for (int i = 0; i < 16; ++i) {
                unsigned    lds = ldsBase + (unsigned)(tid + 256 * i) * 16u;
                const char* g   = gBase + (size_t)(tid + 256 * i) * 16u;
                asm volatile("global_load_async_to_lds_b128 %0, %1, off"
                             :: "v"(lds), "v"(g) : "memory");
            }
            asm volatile("s_wait_asynccnt 0x0" ::: "memory");
        }
        __syncthreads();

        v8f acc[4] = {};
        for (int kk = 0; kk < 512; kk += 32) {
            v16bf a = load_a_frag<512>(Hs, wm * 16, kk, lane);
            const int kt8 = (kk >> 5) * 8;
#pragma unroll
            for (int i = 0; i < 4; ++i) {
                v16bf b = load_b_frag_sw(Us, kt8 + wn * 4 + i, lane);
                acc[i] = __builtin_amdgcn_wmma_f32_16x16x32_bf16(
                    false, a, false, b, (short)0, acc[i], false, false);
            }
        }
        {   // spill transposed: lane's 8 consecutive M -> two b128 stores per acc
            int mr = wm * 16 + ((lane >> 4) << 3);
#pragma unroll
            for (int i = 0; i < 4; ++i) {
                int nc = wn * 64 + (lane & 15) + 16 * i;
                float4 f0 = make_float4(acc[i][0], acc[i][1], acc[i][2], acc[i][3]);
                float4 f1 = make_float4(acc[i][4], acc[i][5], acc[i][6], acc[i][7]);
                *(float4*)&ZsT[nc * 64 + mr]     = f0;
                *(float4*)&ZsT[nc * 64 + mr + 4] = f1;
            }
        }
        __syncthreads();

        for (int i = tid; i < 64 * SLICE; i += 256) {
            int b = i >> 5, j = i & 31;
            size_t zr = ((size_t)b * 1024 + t) * 4096 + colBase;
            float zi = ZsT[(j)      * 64 + b] + Z[zr +        j0 + j];
            float zf = ZsT[(32 + j) * 64 + b] + Z[zr +  512 + j0 + j];
            float zg = ZsT[(64 + j) * 64 + b] + Z[zr + 1024 + j0 + j];
            float zo = ZsT[(96 + j) * 64 + b] + Z[zr + 1536 + j0 + j];
            float ig = 1.f / (1.f + __expf(-zi));
            float fg = 1.f / (1.f + __expf(-zf));
            float gg = tanhf(zg);
            float og = 1.f / (1.f + __expf(-zo));
            float c  = fg * Cs[i] + ig * gg;
            Cs[i] = c;
            float h = og * tanhf(c);
            Hout[(size_t)b * (1024 * 1024) + (size_t)t * 1024 + dirOff + j0 + j] = h;
            hprev[b * 512 + j0 + j] = f32_to_bf16(h);
        }
        grid_sync(sync);
    }
}

// ---------------------------------------------------------------------------
// Host-side orchestration
// ---------------------------------------------------------------------------
extern "C" void kernel_launch(void* const* d_in, const int* in_sizes, int n_in,
                              void* d_out, int out_size, void* d_ws, size_t ws_size,
                              hipStream_t stream)
{
    const float* x   = (const float*)d_in[0];
    const float* W1f = (const float*)d_in[1];
    const float* U1f = (const float*)d_in[2];
    const float* b1f = (const float*)d_in[3];
    const float* W1b = (const float*)d_in[4];
    const float* U1b = (const float*)d_in[5];
    const float* b1b = (const float*)d_in[6];
    const float* W2f = (const float*)d_in[7];
    const float* U2f = (const float*)d_in[8];
    const float* b2f = (const float*)d_in[9];
    const float* W2b = (const float*)d_in[10];
    const float* U2b = (const float*)d_in[11];
    const float* b2b = (const float*)d_in[12];
    const float* Wd  = (const float*)d_in[13];
    const float* bd  = (const float*)d_in[14];

    char* ws = (char*)d_ws;
    size_t off = 0;
    auto alloc = [&](size_t bytes) -> char* {
        char* p = ws + off;
        off += (bytes + 255) & ~(size_t)255;
        return p;
    };

    unsigned*       sync  = (unsigned*)alloc(256);
    unsigned short* hprev = (unsigned short*)alloc((size_t)64 * 512 * 2);

    // Swizzled weight buffers (uints, K/2 * N each)
    unsigned* W1cat = (unsigned*)alloc((size_t)256 * 4096 * 4);   // W1f ‖ W1b
    unsigned* W2cat = (unsigned*)alloc((size_t)512 * 4096 * 4);   // W2f ‖ W2b
    unsigned* U1fp  = (unsigned*)alloc((size_t)256 * 2048 * 4);
    unsigned* U1bp  = (unsigned*)alloc((size_t)256 * 2048 * 4);
    unsigned* U2fp  = (unsigned*)alloc((size_t)256 * 2048 * 4);
    unsigned* U2bp  = (unsigned*)alloc((size_t)256 * 2048 * 4);
    unsigned* Wdp   = (unsigned*)alloc((size_t)512 * 512  * 4);
    float*    bcat1 = (float*)alloc((size_t)4096 * 4);
    float*    bcat2 = (float*)alloc((size_t)4096 * 4);

    float* Zc = (float*)alloc((size_t)65536 * 4096 * 4);  // fused fwd|bwd Z
    float* h1 = (float*)alloc((size_t)65536 * 1024 * 4);
    float* h2 = (float*)alloc((size_t)65536 * 1024 * 4);

    init_sync_kernel<<<1, 1, 0, stream>>>(sync);

    auto cvt = [&](const float* s, unsigned* d, int K2, int Nsrc, int Ntot, int colOff) {
        cvt_swz_kernel<<<(K2 * Nsrc + 255) / 256, 256, 0, stream>>>(s, d, K2, Nsrc, Ntot, colOff);
    };
    cvt(W1f, W1cat, 256, 2048, 4096, 0);
    cvt(W1b, W1cat, 256, 2048, 4096, 2048);
    cvt(W2f, W2cat, 512, 2048, 4096, 0);
    cvt(W2b, W2cat, 512, 2048, 4096, 2048);
    cvt(U1f, U1fp, 256, 2048, 2048, 0);
    cvt(U1b, U1bp, 256, 2048, 2048, 0);
    cvt(U2f, U2fp, 256, 2048, 2048, 0);
    cvt(U2b, U2bp, 256, 2048, 2048, 0);
    cvt(Wd,  Wdp,  512,  512,  512, 0);

    copy_f32_kernel<<<8, 256, 0, stream>>>(b1f, bcat1,        2048);
    copy_f32_kernel<<<8, 256, 0, stream>>>(b1b, bcat1 + 2048, 2048);
    copy_f32_kernel<<<8, 256, 0, stream>>>(b2f, bcat2,        2048);
    copy_f32_kernel<<<8, 256, 0, stream>>>(b2b, bcat2 + 2048, 2048);

    const dim3 blk(256);
    // Layer 1 fused projection: Zc = x @ [W1f‖W1b] + [b1f‖b1b]  (N=4096, K=512)
    gemm_bias<false><<<dim3(32, 512), blk, 0, stream>>>(x, nullptr, W1cat, bcat1, Zc, 4096, 512);
    // Layer 1 recurrences -> h1
    lstm_rec<<<dim3(NBLK), blk, 0, stream>>>(Zc, U1fp, h1, hprev, sync, 0, 0,    0);
    lstm_rec<<<dim3(NBLK), blk, 0, stream>>>(Zc, U1bp, h1, hprev, sync, 1, 512, 2048);
    // Layer 2 fused projection: Zc = h1 @ [W2f‖W2b] + [b2f‖b2b]  (N=4096, K=1024)
    gemm_bias<false><<<dim3(32, 512), blk, 0, stream>>>(h1, nullptr, W2cat, bcat2, Zc, 4096, 1024);
    // Layer 2 recurrences -> h2
    lstm_rec<<<dim3(NBLK), blk, 0, stream>>>(Zc, U2fp, h2, hprev, sync, 0, 0,    0);
    lstm_rec<<<dim3(NBLK), blk, 0, stream>>>(Zc, U2bp, h2, hprev, sync, 1, 512, 2048);
    // Dense head with fused residual: out = (h1 + h2) @ Wd + bd  (N=512, K=1024)
    gemm_bias<true><<<dim3(4, 512), blk, 0, stream>>>(h1, h2, Wdp, bd, (float*)d_out, 512, 1024);
}
